// CAM_Module_35588099015599
// MI455X (gfx1250) — matrix operand order repaired and checked by reading, not compiled, not run on hardware
//
#include <hip/hip_runtime.h>
#include <hip/hip_bf16.h>

#define B_ 8
#define C_ 512
#define N_ 9216   // 96*96

typedef __attribute__((ext_vector_type(16))) __bf16 v16bf;
typedef __attribute__((ext_vector_type(8)))  float  v8f;

typedef __attribute__((address_space(1))) int* gint_p;   // global AS
typedef __attribute__((address_space(3))) int* lint_p;   // LDS AS

union FragBF { v16bf v; uint4 u[2]; };

#define WMMA_BF16(a, b, c) \
    __builtin_amdgcn_wmma_f32_16x16x32_bf16(false, (a).v, false, (b).v, (short)0, (c), false, false)

// ---------------------------------------------------------------------------
// x3[b][n] = max_c x[b][c][n] + mean_c x[b][c][n]
// ---------------------------------------------------------------------------
__global__ void pool_kernel(const float* __restrict__ x, float* __restrict__ x3) {
    int idx = blockIdx.x * blockDim.x + threadIdx.x;      // b*N + n
    int b = idx / N_;
    int n = idx - b * N_;
    const float* p = x + (size_t)b * C_ * N_ + n;
    float mx = -3.402823466e38f, sum = 0.0f;
#pragma unroll 4
    for (int c = 0; c < C_; ++c) {
        float v = p[(size_t)c * N_];
        mx = fmaxf(mx, v);
        sum += v;
    }
    x3[idx] = mx + sum * (1.0f / (float)C_);
}

// ---------------------------------------------------------------------------
// qh = bf16(x), elementwise (float4 -> 4x bf16)
// ---------------------------------------------------------------------------
__global__ void conv_q_kernel(const float* __restrict__ x, __bf16* __restrict__ qh) {
    size_t i = ((size_t)blockIdx.x * blockDim.x + threadIdx.x) * 4;
    float4 v = *(const float4*)(x + i);
    union { __bf16 h[4]; uint2 u; } o;
    o.h[0] = (__bf16)v.x; o.h[1] = (__bf16)v.y;
    o.h[2] = (__bf16)v.z; o.h[3] = (__bf16)v.w;
    *(uint2*)(qh + i) = o.u;
}

// ---------------------------------------------------------------------------
// pvT[b][n][c] = bf16( x[b][c][n] * x3[b][n] )   (LDS 32x32 tile transpose)
// Uses gfx1250 async global->LDS copy (ASYNCcnt) when the builtin exists.
// ---------------------------------------------------------------------------
__global__ void pv_transpose_kernel(const float* __restrict__ x,
                                    const float* __restrict__ x3,
                                    __bf16* __restrict__ pvT) {
    __shared__ float tile[32][33];
    int b = blockIdx.z;
    int cBase = blockIdx.y * 32;
    int nBase = blockIdx.x * 32;
    int tx = threadIdx.x, ty = threadIdx.y;
    const float* gsrc = x + ((size_t)(b * C_ + cBase + ty)) * N_ + nBase + tx;
#if __has_builtin(__builtin_amdgcn_global_load_async_to_lds_b32)
    // async DMA: LDS[per-lane dst] = MEM[per-lane src]; tracked by ASYNCcnt
    __builtin_amdgcn_global_load_async_to_lds_b32(
        (gint_p)const_cast<float*>(gsrc),
        (lint_p)&tile[ty][tx], 0, 0);
#if __has_builtin(__builtin_amdgcn_s_wait_asynccnt)
    __builtin_amdgcn_s_wait_asynccnt(0);
#else
    asm volatile("s_wait_asynccnt 0" ::: "memory");
#endif
    __syncthreads();
#else
    tile[ty][tx] = *gsrc;
    __syncthreads();
#endif
    int n = nBase + ty;
    float s = x3[b * N_ + n];
    pvT[((size_t)b * N_ + n) * C_ + cBase + tx] = (__bf16)(tile[tx][ty] * s);
}

// ---------------------------------------------------------------------------
// C/D tile store helpers.  Element v of a v8f accumulator is
// (row = r + 8*half + v, col = cCol + m) of the 16x16 tile.
// ---------------------------------------------------------------------------
__device__ __forceinline__ void store_tile(float* __restrict__ energy, int b, int r, int cCol,
                                           const v8f& acc, int m, int half) {
    float* o = energy + ((size_t)b * C_ + r + 8 * half) * C_ + cCol + m;
#pragma unroll
    for (int v = 0; v < 8; ++v) o[(size_t)v * C_] = acc[v];
}
__device__ __forceinline__ void mirror_tile(float* __restrict__ energy, int b, int r, int cCol,
                                            const v8f& acc, int m, int half) {
    // mirrored: energy[cCol + m][r + 8*half + v]  (contiguous per lane)
    float* o = energy + ((size_t)b * C_ + cCol + m) * C_ + r + 8 * half;
#pragma unroll
    for (int v = 0; v < 8; ++v) o[v] = acc[v];
}

// ---------------------------------------------------------------------------
// energy[b] = q q^T : [512 x 9216]*[9216 x 512], bf16 WMMA, f32 accum.
// 4 waves/block, each wave a 32x32 register tile (2x2 WMMA) -> 64x64 block.
// Symmetric output: only lower-triangle blocks computed; mirror-stored.
// ---------------------------------------------------------------------------
__global__ void gram_kernel(const __bf16* __restrict__ qh, float* __restrict__ energy) {
    int dTile = blockIdx.x;           // 64-wide column tile
    int cTile = blockIdx.y;           // 64-wide row tile
    if (cTile < dTile) return;        // symmetry: lower triangle only
    int b    = blockIdx.z;
    int wave = threadIdx.x >> 5;
    int lane = threadIdx.x & 31;
    int m    = lane & 15;
    int half = lane >> 4;
    int cTop = cTile * 64 + (wave >> 1) * 32;
    int dTop = dTile * 64 + (wave & 1) * 32;

    const __bf16* A0 = qh + ((size_t)b * C_ + cTop + m) * N_;
    const __bf16* A1 = A0 + (size_t)16 * N_;
    const __bf16* B0 = qh + ((size_t)b * C_ + dTop + m) * N_;
    const __bf16* B1 = B0 + (size_t)16 * N_;

    v8f acc00 = {}, acc01 = {}, acc10 = {}, acc11 = {};
    for (int k = 0; k < N_; k += 32) {
        FragBF a0, a1, b0, b1;
        a0.u[0] = *(const uint4*)(A0 + k + 8 * half);
        a0.u[1] = *(const uint4*)(A0 + k + 16 + 8 * half);
        a1.u[0] = *(const uint4*)(A1 + k + 8 * half);
        a1.u[1] = *(const uint4*)(A1 + k + 16 + 8 * half);
        b0.u[0] = *(const uint4*)(B0 + k + 8 * half);
        b0.u[1] = *(const uint4*)(B0 + k + 16 + 8 * half);
        b1.u[0] = *(const uint4*)(B1 + k + 8 * half);
        b1.u[1] = *(const uint4*)(B1 + k + 16 + 8 * half);
        acc00 = WMMA_BF16(a0, b0, acc00);
        acc01 = WMMA_BF16(a0, b1, acc01);
        acc10 = WMMA_BF16(a1, b0, acc10);
        acc11 = WMMA_BF16(a1, b1, acc11);
    }
    store_tile(energy, b, cTop,      dTop,      acc00, m, half);
    store_tile(energy, b, cTop,      dTop + 16, acc01, m, half);
    store_tile(energy, b, cTop + 16, dTop,      acc10, m, half);
    store_tile(energy, b, cTop + 16, dTop + 16, acc11, m, half);
    if (cTile != dTile) {
        mirror_tile(energy, b, cTop,      dTop,      acc00, m, half);
        mirror_tile(energy, b, cTop,      dTop + 16, acc01, m, half);
        mirror_tile(energy, b, cTop + 16, dTop,      acc10, m, half);
        mirror_tile(energy, b, cTop + 16, dTop + 16, acc11, m, half);
    }
}

// ---------------------------------------------------------------------------
// attention[b][c][d] = softmax_d( rowmax - energy ) == exp(rowmin - e)/sum
// ---------------------------------------------------------------------------
__global__ void softmax_kernel(const float* __restrict__ energy, __bf16* __restrict__ attn) {
    __shared__ float red[C_];
    int row = blockIdx.x;                 // b*C + c
    int d = threadIdx.x;
    float e = energy[(size_t)row * C_ + d];
    red[d] = e;
    __syncthreads();
    for (int s = 256; s > 0; s >>= 1) {
        if (d < s) red[d] = fminf(red[d], red[d + s]);
        __syncthreads();
    }
    float mn = red[0];
    __syncthreads();
    float p = __expf(mn - e);
    red[d] = p;
    __syncthreads();
    for (int s = 256; s > 0; s >>= 1) {
        if (d < s) red[d] += red[d + s];
        __syncthreads();
    }
    attn[(size_t)row * C_ + d] = (__bf16)(p / red[0]);
}

// ---------------------------------------------------------------------------
// out = gamma * (attention @ pv) + x : [512 x 512]*[512 x 9216]
// Same 2x2-per-wave / 64x64-per-block tiling; fused epilogue.
// ---------------------------------------------------------------------------
__global__ void out_kernel(const __bf16* __restrict__ attn, const __bf16* __restrict__ pvT,
                           const float* __restrict__ x, const float* __restrict__ gamma,
                           float* __restrict__ out) {
    int b    = blockIdx.z;
    int wave = threadIdx.x >> 5;
    int lane = threadIdx.x & 31;
    int m    = lane & 15;
    int half = lane >> 4;
    int cTop = blockIdx.y * 64 + (wave >> 1) * 32;
    int nTop = blockIdx.x * 64 + (wave & 1) * 32;

    const __bf16* A0 = attn + ((size_t)b * C_ + cTop + m) * C_;
    const __bf16* A1 = A0 + (size_t)16 * C_;
    const __bf16* B0 = pvT + ((size_t)b * N_ + nTop + m) * C_;
    const __bf16* B1 = B0 + (size_t)16 * C_;

    v8f acc00 = {}, acc01 = {}, acc10 = {}, acc11 = {};
#pragma unroll 4
    for (int k = 0; k < C_; k += 32) {
        FragBF a0, a1, b0, b1;
        a0.u[0] = *(const uint4*)(A0 + k + 8 * half);
        a0.u[1] = *(const uint4*)(A0 + k + 16 + 8 * half);
        a1.u[0] = *(const uint4*)(A1 + k + 8 * half);
        a1.u[1] = *(const uint4*)(A1 + k + 16 + 8 * half);
        b0.u[0] = *(const uint4*)(B0 + k + 8 * half);
        b0.u[1] = *(const uint4*)(B0 + k + 16 + 8 * half);
        b1.u[0] = *(const uint4*)(B1 + k + 8 * half);
        b1.u[1] = *(const uint4*)(B1 + k + 16 + 8 * half);
        acc00 = WMMA_BF16(a0, b0, acc00);
        acc01 = WMMA_BF16(a0, b1, acc01);
        acc10 = WMMA_BF16(a1, b0, acc10);
        acc11 = WMMA_BF16(a1, b1, acc11);
    }
    float g = gamma[0];
#pragma unroll
    for (int t = 0; t < 4; ++t) {
        int r  = cTop + (t >> 1) * 16;          // output row block
        int nc = nTop + (t & 1) * 16;           // output col block
        const v8f& a = (t == 0) ? acc00 : (t == 1) ? acc01 : (t == 2) ? acc10 : acc11;
        size_t base = ((size_t)b * C_ + r + 8 * half) * N_ + nc + m;
#pragma unroll
        for (int v = 0; v < 8; ++v) {
            size_t idx = base + (size_t)v * N_;
            out[idx] = g * a[v] + x[idx];
        }
    }
}

// ---------------------------------------------------------------------------
extern "C" void kernel_launch(void* const* d_in, const int* in_sizes, int n_in,
                              void* d_out, int out_size, void* d_ws, size_t ws_size,
                              hipStream_t stream) {
    const float* x     = (const float*)d_in[0];
    const float* gamma = (const float*)d_in[1];
    float*       out   = (float*)d_out;

    char* ws = (char*)d_ws;
    const size_t qh_bytes     = (size_t)B_ * C_ * N_ * 2;   // 75.5 MB
    const size_t energy_bytes = (size_t)B_ * C_ * C_ * 4;   // 8 MB
    const size_t attn_bytes   = (size_t)B_ * C_ * C_ * 2;   // 4 MB

    __bf16* qh     = (__bf16*)(ws);
    __bf16* pvT    = (__bf16*)(ws + qh_bytes);
    float*  energy = (float*) (ws + 2 * qh_bytes);
    __bf16* attn   = (__bf16*)(ws + 2 * qh_bytes + energy_bytes);
    float*  x3     = (float*) (ws + 2 * qh_bytes + energy_bytes + attn_bytes);

    pool_kernel<<<dim3((B_ * N_) / 256), dim3(256), 0, stream>>>(x, x3);

    conv_q_kernel<<<dim3((unsigned)((size_t)B_ * C_ * N_ / 4 / 256)), dim3(256), 0, stream>>>(x, qh);

    pv_transpose_kernel<<<dim3(N_ / 32, C_ / 32, B_), dim3(32, 32), 0, stream>>>(x, x3, pvT);

    gram_kernel<<<dim3(C_ / 64, C_ / 64, B_), dim3(128), 0, stream>>>(qh, energy);

    softmax_kernel<<<dim3(B_ * C_), dim3(C_), 0, stream>>>(energy, attn);

    out_kernel<<<dim3(N_ / 64, C_ / 64, B_), dim3(128), 0, stream>>>(attn, pvT, x, gamma, out);
}
